// MultiSpanRNN_57947698757786
// MI455X (gfx1250) — matrix operand order repaired and checked by reading, not compiled, not run on hardware
//
#include <hip/hip_runtime.h>

// ---------------------------------------------------------------------------
// MultiSpanRNN on gfx1250: bf16 WMMA (v_wmma_f32_16x16x32_bf16) everywhere.
// Pipeline per launch (all on `stream`):
//   0) zero barrier words
//   1) convert x + all weight matrices fp32 -> bf16 in ws
//   2) Gi0[m] = x @ W_ih0[m].T + b_ih0[m]      (parallel WMMA GEMM, 6x)
//   3) persistent sequential kernel: 3 spans x 8 WGs, 256 steps each,
//      4 WMMA phases per step with span-wide sense-reversing barriers
//   4) out = cat @ lin_W.T + lin_b             (parallel WMMA GEMM)
// ---------------------------------------------------------------------------

typedef __bf16 bf16;
typedef __attribute__((ext_vector_type(16))) __bf16 v16bf;
typedef __attribute__((ext_vector_type(8)))  float  v8f;
typedef __attribute__((ext_vector_type(4)))  float  f32x4;

namespace {
constexpr int kB = 32;      // batch
constexpr int kT = 256;     // seq len
constexpr int kF = 1536;    // features (= 3*kH)
constexpr int kH = 512;     // hidden
constexpr int kNWG = 8;     // workgroups per span
constexpr int kBH = kB * kH;
constexpr int kSlice = kH / kNWG;   // 64 columns of H per WG
}

union Frag16 { v16bf v; f32x4 q[2]; };

// A-matrix fragment (16x32 bf16, row-major, leading dim ld).
// lane L: row = m0 + (L&15); elems 0..7 = K[kb..kb+7], 8..15 = K[kb+16..kb+23],
// kb = (L>=16) ? 8 : 0.
__device__ inline v16bf ldA(const bf16* base, int ld, int lane) {
  const bf16* p = base + (size_t)(lane & 15) * ld + ((lane >> 4) << 3);
  Frag16 f;
  f.q[0] = *(const f32x4*)(p);
  f.q[1] = *(const f32x4*)(p + 16);
  return f.v;
}

// B-matrix fragment for Y = A @ W^T with W row-major (N x K): B[k,n] = W[n,k].
// lane L: col n = n0 + (L&15); elems 0..7 = K[kb..kb+7], 8..15 = K[kb+16..kb+23], kb = (L>=16) ? 8 : 0.
__device__ inline v16bf ldB(const bf16* base, int ld, int lane) {
  const bf16* p = base + (size_t)(lane & 15) * ld + ((lane >> 4) << 3);
  Frag16 f;
  f.q[0] = *(const f32x4*)(p);
  f.q[1] = *(const f32x4*)(p + 16);
  return f.v;
}

__device__ inline v8f wmma_bf(v16bf a, v16bf b, v8f c) {
  return __builtin_amdgcn_wmma_f32_16x16x32_bf16(
      /*neg_a=*/false, a, /*neg_b=*/false, b,
      /*c_mod=*/(short)0, c, /*reuse_a=*/false, /*reuse_b=*/false);
}

__device__ inline float sigm(float x) { return 1.0f / (1.0f + __expf(-x)); }

// Sense-reversing span-wide barrier over nwg workgroups. bar[0]=count, bar[1]=gen.
__device__ inline void span_barrier(unsigned* bar, int nwg) {
  __syncthreads();
  if (threadIdx.x == 0) {
    unsigned* cnt = bar;
    unsigned* gen = bar + 1;
    unsigned g = __hip_atomic_load(gen, __ATOMIC_ACQUIRE, __HIP_MEMORY_SCOPE_AGENT);
    unsigned a = __hip_atomic_fetch_add(cnt, 1u, __ATOMIC_ACQ_REL, __HIP_MEMORY_SCOPE_AGENT);
    if (a + 1u == (unsigned)nwg) {
      __hip_atomic_store(cnt, 0u, __ATOMIC_RELAXED, __HIP_MEMORY_SCOPE_AGENT);
      __hip_atomic_fetch_add(gen, 1u, __ATOMIC_ACQ_REL, __HIP_MEMORY_SCOPE_AGENT);
    } else {
      while (__hip_atomic_load(gen, __ATOMIC_ACQUIRE, __HIP_MEMORY_SCOPE_AGENT) == g)
        __builtin_amdgcn_s_sleep(2);
    }
  }
  __syncthreads();
}

// ---------------------------------------------------------------------------
__global__ void init_bars_kernel(unsigned* bars) {
  if (threadIdx.x < 48) bars[threadIdx.x] = 0u;
}

__global__ void cvt_f32_to_bf16(const float* __restrict__ s, bf16* __restrict__ d, long n) {
  long i = (long)blockIdx.x * blockDim.x + threadIdx.x;
  long stride = (long)gridDim.x * blockDim.x;
  for (; i < n; i += stride) d[i] = (bf16)s[i];
}

// ---------------------------------------------------------------------------
// Y[M,N] = A[M,K] @ W[N,K]^T + bias[N], bf16 inputs, f32 accumulate.
// block = 256 threads (8 waves); block tile = 32(M) x 128(N); grid.z batches W/bias/Y.
template <bool OUTBF>
__global__ __launch_bounds__(256) void gemm_awt_kernel(
    const bf16* __restrict__ A, const bf16* __restrict__ W,
    const float* __restrict__ bias, void* __restrict__ Yv,
    int M, int N, int K, long wz, long bz, long yz) {
  const int z = blockIdx.z;
  W += (size_t)z * wz;
  bias += (size_t)z * bz;
  const int lane = threadIdx.x & 31;
  const int wv = threadIdx.x >> 5;
  const int m0 = blockIdx.y * 32;
  const int n0 = blockIdx.x * 128 + wv * 16;
  const bf16* Ar0 = A + (size_t)m0 * K;
  const bf16* Ar1 = A + (size_t)(m0 + 16) * K;
  const bf16* Wr = W + (size_t)n0 * K;
  v8f acc0 = {}, acc1 = {};
  for (int k = 0; k < K; k += 32) {
    if (k + 32 < K) __builtin_prefetch(Wr + k + 32, 0, 1);
    v16bf bfr = ldB(Wr + k, K, lane);
    v16bf a0 = ldA(Ar0 + k, K, lane);
    v16bf a1 = ldA(Ar1 + k, K, lane);
    acc0 = wmma_bf(a0, bfr, acc0);
    acc1 = wmma_bf(a1, bfr, acc1);
  }
  const int col = n0 + (lane & 15);
  const int rb = (lane >> 4) << 3;
  const float bb = bias[col];
  if (OUTBF) {
    bf16* Y = (bf16*)Yv + (size_t)z * yz;
    for (int r = 0; r < 8; ++r) {
      Y[(size_t)(m0 + rb + r) * N + col] = (bf16)(acc0[r] + bb);
      Y[(size_t)(m0 + 16 + rb + r) * N + col] = (bf16)(acc1[r] + bb);
    }
  } else {
    float* Y = (float*)Yv + (size_t)z * yz;
    for (int r = 0; r < 8; ++r) {
      Y[(size_t)(m0 + rb + r) * N + col] = acc0[r] + bb;
      Y[(size_t)(m0 + 16 + rb + r) * N + col] = acc1[r] + bb;
    }
  }
}

// ---------------------------------------------------------------------------
struct SeqArgs {
  const bf16 *Whh0, *Wih1, *Whh1, *gateW;  // bf16 weights (6-GRU stacks / gate)
  const bf16 *Gi;                          // [6][B*T][3H] precomputed x-projection
  const float *b_hh0, *b_ih1, *b_hh1, *gate_b, *h0;
  bf16* cat;                               // [B*T][F] bf16 output staging
  float* stf;                              // per-span f32 state, stride 12*kBH
  bf16* stb;                               // per-span bf16 state, stride 10*kBH
  unsigned* bars;                          // [3][16]
};

__global__ __launch_bounds__(256) void span_seq_kernel(SeqArgs a) {
  const int span = blockIdx.x / kNWG;
  const int wg = blockIdx.x % kNWG;
  const int tid = threadIdx.x;
  const int lane = tid & 31;
  const int wv = tid >> 5;
  const int spanlen = (span == 0) ? 1 : ((span == 1) ? 3 : 5);

  float* h1 = a.stf + (size_t)span * 12 * kBH;   // [layer][B][H]
  float* h2 = h1 + 2 * kBH;
  float* hp1 = h1 + 4 * kBH;
  float* hp2 = h1 + 6 * kBH;
  float* tn = h1 + 8 * kBH;                      // [stream][layer][B][H]
  bf16* h1b = a.stb + (size_t)span * 10 * kBH;   // [layer][B][H]
  bf16* h2b = h1b + 2 * kBH;
  bf16* h0nb = h1b + 4 * kBH;                    // [stream][B][H]
  bf16* comb = h1b + 6 * kBH;                    // [layer][B][2H]
  unsigned* bar = a.bars + span * 16;

  const int jbase = wg * kSlice;
  const int m0 = (wv >> 2) * 16;                 // batch-tile (B=32 -> 2 tiles)
  const int j0 = jbase + (wv & 3) * 16;          // per-wave 16 cols of the slice
  const int colw = j0 + (lane & 15);
  const int rb = (lane >> 4) << 3;

  // init: h1,h2,hp1,hp2 <- h0 ; bf16 mirrors
  for (int e = tid; e < 2 * 2 * kB * kSlice; e += 256) {
    int c = e % kSlice;
    int r2 = e / kSlice;
    int b = r2 % kB;
    int r3 = r2 / kB;
    int l = r3 & 1;
    int s = r3 >> 1;
    float v = a.h0[(((size_t)s * 2 + l) * kB + b) * kH + jbase + c];
    size_t o = ((size_t)l * kB + b) * kH + jbase + c;
    if (s == 0) { h1[o] = v; hp1[o] = v; h1b[o] = (bf16)v; }
    else        { h2[o] = v; hp2[o] = v; h2b[o] = (bf16)v; }
  }
  span_barrier(bar, kNWG);

  for (int t = 0; t < kT; ++t) {
    // static span plan: segment, reset, original time index, skip/gate flags
    int rem = t, seg = 0;
    for (seg = 0; seg < spanlen; ++seg) {
      int L = (kT - seg + spanlen - 1) / spanlen;
      if (rem < L) break;
      rem -= L;
    }
    const bool rs = (rem == 0);
    const int orig = seg + rem * spanlen;
    const bool sk = (((t + 1) % 6) == 0);
    const bool gt = (((t + 1) % 4) != 0);

    if (rs) {  // reset hidden (not hp) at sub-span starts; uniform across span
      for (int e = tid; e < 2 * 2 * kB * kSlice; e += 256) {
        int c = e % kSlice;
        int r2 = e / kSlice;
        int b = r2 % kB;
        int r3 = r2 / kB;
        int l = r3 & 1;
        int s = r3 >> 1;
        float v = a.h0[(((size_t)s * 2 + l) * kB + b) * kH + jbase + c];
        size_t o = ((size_t)l * kB + b) * kH + jbase + c;
        if (s == 0) { h1[o] = v; h1b[o] = (bf16)v; }
        else        { h2[o] = v; h2b[o] = (bf16)v; }
      }
      span_barrier(bar, kNWG);
    }

    // ---- Phase A: layer0 (gh = h @ W_hh0^T), both streams ----
    for (int s = 0; s < 2; ++s) {
      const int m = 2 * span + s;
      const bf16* hb = (s ? h2b : h1b);            // layer0 prev hidden (bf16)
      const float* hf = (s ? h2 : h1);             // layer0 prev hidden (f32)
      const bf16* Wh = a.Whh0 + (size_t)m * 3 * kH * kH;  // (3H x H) row-major
      v8f acc0 = {}, acc1 = {}, acc2 = {};
      for (int k = 0; k < kH; k += 32) {
        v16bf af = ldA(hb + (size_t)m0 * kH + k, kH, lane);
        v16bf b0 = ldB(Wh + (size_t)(0 * kH + j0) * kH + k, kH, lane);
        v16bf b1 = ldB(Wh + (size_t)(1 * kH + j0) * kH + k, kH, lane);
        v16bf b2 = ldB(Wh + (size_t)(2 * kH + j0) * kH + k, kH, lane);
        acc0 = wmma_bf(af, b0, acc0);
        acc1 = wmma_bf(af, b1, acc1);
        acc2 = wmma_bf(af, b2, acc2);
      }
      const bf16* gi = a.Gi + (size_t)m * kB * kT * (3 * kH);
      const float* bh = a.b_hh0 + (size_t)m * 3 * kH;
      for (int r = 0; r < 8; ++r) {
        int b = m0 + rb + r;
        size_t gir = ((size_t)b * kT + orig) * (3 * kH) + colw;
        float i_r = (float)gi[gir];
        float i_z = (float)gi[gir + kH];
        float i_n = (float)gi[gir + 2 * kH];
        float h_r = acc0[r] + bh[colw];
        float h_z = acc1[r] + bh[kH + colw];
        float h_n = acc2[r] + bh[2 * kH + colw];
        float rr = sigm(i_r + h_r);
        float zz = sigm(i_z + h_z);
        float nn = tanhf(i_n + rr * h_n);
        float hpv = hf[(size_t)b * kH + colw];
        float v = (1.0f - zz) * nn + zz * hpv;
        tn[((size_t)(s * 2 + 0) * kB + b) * kH + colw] = v;
        h0nb[((size_t)s * kB + b) * kH + colw] = (bf16)v;
      }
    }
    span_barrier(bar, kNWG);

    // ---- Phase B: layer1 (gi = h0n @ W_ih1^T, gh = h @ W_hh1^T) ----
    for (int s = 0; s < 2; ++s) {
      const int m = 2 * span + s;
      const bf16* xb = h0nb + (size_t)s * kBH;
      const bf16* hb = (s ? h2b : h1b) + kBH;      // layer1 prev hidden (bf16)
      const float* hf = (s ? h2 : h1) + kBH;
      const bf16* Wi = a.Wih1 + (size_t)m * 3 * kH * kH;
      const bf16* Wh = a.Whh1 + (size_t)m * 3 * kH * kH;
      v8f accR = {}, accZ = {}, accNi = {}, accNh = {};
      for (int k = 0; k < kH; k += 32) {
        v16bf ai = ldA(xb + (size_t)m0 * kH + k, kH, lane);
        v16bf ah = ldA(hb + (size_t)m0 * kH + k, kH, lane);
        v16bf bi0 = ldB(Wi + (size_t)(0 * kH + j0) * kH + k, kH, lane);
        v16bf bi1 = ldB(Wi + (size_t)(1 * kH + j0) * kH + k, kH, lane);
        v16bf bi2 = ldB(Wi + (size_t)(2 * kH + j0) * kH + k, kH, lane);
        v16bf bh0 = ldB(Wh + (size_t)(0 * kH + j0) * kH + k, kH, lane);
        v16bf bh1 = ldB(Wh + (size_t)(1 * kH + j0) * kH + k, kH, lane);
        v16bf bh2 = ldB(Wh + (size_t)(2 * kH + j0) * kH + k, kH, lane);
        accR = wmma_bf(ai, bi0, accR);
        accR = wmma_bf(ah, bh0, accR);
        accZ = wmma_bf(ai, bi1, accZ);
        accZ = wmma_bf(ah, bh1, accZ);
        accNi = wmma_bf(ai, bi2, accNi);   // i_n part (r * h_n needs h_n separate)
        accNh = wmma_bf(ah, bh2, accNh);   // h_n part
      }
      const float* bi = a.b_ih1 + (size_t)m * 3 * kH;
      const float* bh = a.b_hh1 + (size_t)m * 3 * kH;
      for (int r = 0; r < 8; ++r) {
        int b = m0 + rb + r;
        float rr = sigm(accR[r] + bi[colw] + bh[colw]);
        float zz = sigm(accZ[r] + bi[kH + colw] + bh[kH + colw]);
        float h_n = accNh[r] + bh[2 * kH + colw];
        float nn = tanhf(accNi[r] + bi[2 * kH + colw] + rr * h_n);
        float hpv = hf[(size_t)b * kH + colw];
        float v = (1.0f - zz) * nn + zz * hpv;
        tn[((size_t)(s * 2 + 1) * kB + b) * kH + colw] = v;
      }
    }
    span_barrier(bar, kNWG);

    // ---- Phase C: residual skip + build comb = [h1n, h2n] per layer (bf16) ----
    for (int e = tid; e < 2 * kB * kSlice; e += 256) {
      int c = e % kSlice;
      int r2 = e / kSlice;
      int b = r2 % kB;
      int l = r2 / kB;
      int col = jbase + c;
      size_t o = ((size_t)l * kB + b) * kH + col;
      size_t o1 = ((size_t)(0 * 2 + l) * kB + b) * kH + col;
      size_t o2 = ((size_t)(1 * 2 + l) * kB + b) * kH + col;
      float v1 = tn[o1], v2 = tn[o2];
      if (sk) {
        v1 += hp1[o]; hp1[o] = v1; tn[o1] = v1;
        v2 += hp2[o]; hp2[o] = v2; tn[o2] = v2;
      }
      comb[((size_t)l * kB + b) * (2 * kH) + col] = (bf16)v1;
      comb[((size_t)l * kB + b) * (2 * kH) + kH + col] = (bf16)v2;
    }
    span_barrier(bar, kNWG);

    // ---- Phase D: gate GEMM + gated fusion + state update + output ----
    for (int l = 0; l < 2; ++l) {
      const bf16* A = comb + (size_t)l * kB * 2 * kH;
      v8f acc = {};
      for (int k = 0; k < 2 * kH; k += 32) {
        v16bf af = ldA(A + (size_t)m0 * (2 * kH) + k, 2 * kH, lane);
        v16bf bg = ldB(a.gateW + (size_t)j0 * (2 * kH) + k, 2 * kH, lane);
        acc = wmma_bf(af, bg, acc);
      }
      for (int r = 0; r < 8; ++r) {
        int b = m0 + rb + r;
        size_t o = ((size_t)l * kB + b) * kH + colw;
        float v1 = tn[((size_t)(0 * 2 + l) * kB + b) * kH + colw];
        float v2 = tn[((size_t)(1 * 2 + l) * kB + b) * kH + colw];
        float g = sigm(acc[r] + a.gate_b[colw]);
        float nh = gt ? (g * v1 + (1.0f - g) * v2) : v1;
        h1[o] = nh; h1b[o] = (bf16)nh;
        h2[o] = v2; h2b[o] = (bf16)v2;
        if (l == 1)
          a.cat[((size_t)b * kT + orig) * kF + span * kH + colw] = (bf16)nh;
      }
    }
    span_barrier(bar, kNWG);
  }
}

// ---------------------------------------------------------------------------
extern "C" void kernel_launch(void* const* d_in, const int* in_sizes, int n_in,
                              void* d_out, int out_size, void* d_ws, size_t ws_size,
                              hipStream_t stream) {
  (void)in_sizes; (void)n_in; (void)out_size; (void)ws_size;
  const float* x      = (const float*)d_in[0];
  const float* W_ih0  = (const float*)d_in[1];
  const float* W_hh0  = (const float*)d_in[2];
  const float* b_ih0  = (const float*)d_in[3];
  const float* b_hh0  = (const float*)d_in[4];
  const float* W_ih1  = (const float*)d_in[5];
  const float* W_hh1  = (const float*)d_in[6];
  const float* b_ih1  = (const float*)d_in[7];
  const float* b_hh1  = (const float*)d_in[8];
  const float* h0     = (const float*)d_in[9];
  const float* gate_W = (const float*)d_in[10];
  const float* gate_b = (const float*)d_in[11];
  const float* lin_W  = (const float*)d_in[12];
  const float* lin_b  = (const float*)d_in[13];

  char* ws = (char*)d_ws;
  size_t off = 0;
  auto take = [&](size_t bytes) {
    char* p = ws + off;
    off = (off + bytes + 255) & ~(size_t)255;
    return p;
  };
  const size_t BT = (size_t)kB * kT;                       // 8192
  bf16* xb     = (bf16*)take(BT * kF * 2);
  bf16* Wih0b  = (bf16*)take((size_t)6 * 3 * kH * kF * 2);
  bf16* Whh0b  = (bf16*)take((size_t)6 * 3 * kH * kH * 2);
  bf16* Wih1b  = (bf16*)take((size_t)6 * 3 * kH * kH * 2);
  bf16* Whh1b  = (bf16*)take((size_t)6 * 3 * kH * kH * 2);
  bf16* gateWb = (bf16*)take((size_t)kH * 2 * kH * 2);
  bf16* linWb  = (bf16*)take((size_t)kF * kF * 2);
  bf16* Gib    = (bf16*)take((size_t)6 * BT * 3 * kH * 2);
  bf16* catb   = (bf16*)take(BT * kF * 2);
  float* stf   = (float*)take((size_t)3 * 12 * kBH * 4);
  bf16* stb    = (bf16*)take((size_t)3 * 10 * kBH * 2);
  unsigned* bars = (unsigned*)take(3 * 16 * 4);

  init_bars_kernel<<<1, 64, 0, stream>>>(bars);

  auto cvt = [&](const float* s, bf16* d, size_t n) {
    long blocks = (long)((n + 256 * 8 - 1) / (256 * 8));
    if (blocks > 2048) blocks = 2048;
    if (blocks < 1) blocks = 1;
    cvt_f32_to_bf16<<<(int)blocks, 256, 0, stream>>>(s, d, (long)n);
  };
  cvt(x, xb, BT * kF);
  cvt(W_ih0, Wih0b, (size_t)6 * 3 * kH * kF);
  cvt(W_hh0, Whh0b, (size_t)6 * 3 * kH * kH);
  cvt(W_ih1, Wih1b, (size_t)6 * 3 * kH * kH);
  cvt(W_hh1, Whh1b, (size_t)6 * 3 * kH * kH);
  cvt(gate_W, gateWb, (size_t)kH * 2 * kH);
  cvt(lin_W, linWb, (size_t)kF * kF);

  // Gi0[m] = x @ W_ih0[m]^T + b_ih0[m]   -> bf16   (M=8192, N=1536, K=1536, z=6)
  gemm_awt_kernel<true><<<dim3((3 * kH) / 128, (int)(BT / 32), 6), 256, 0, stream>>>(
      xb, Wih0b, b_ih0, Gib, (int)BT, 3 * kH, kF,
      (long)3 * kH * kF, (long)3 * kH, (long)(BT * 3 * kH));

  SeqArgs sa;
  sa.Whh0 = Whh0b; sa.Wih1 = Wih1b; sa.Whh1 = Whh1b; sa.gateW = gateWb;
  sa.Gi = Gib;
  sa.b_hh0 = b_hh0; sa.b_ih1 = b_ih1; sa.b_hh1 = b_hh1;
  sa.gate_b = gate_b; sa.h0 = h0;
  sa.cat = catb; sa.stf = stf; sa.stb = stb; sa.bars = bars;
  span_seq_kernel<<<3 * kNWG, 256, 0, stream>>>(sa);

  // out = cat @ lin_W^T + lin_b  -> f32  (M=8192, N=1536, K=1536)
  gemm_awt_kernel<false><<<dim3(kF / 128, (int)(BT / 32), 1), 256, 0, stream>>>(
      catb, linWb, lin_b, d_out, (int)BT, kF, kF, 0, 0, 0);
}
